// SelfAttention_35837207117933
// MI455X (gfx1250) — matrix operand order repaired and checked by reading, not compiled, not run on hardware
//
#include <hip/hip_runtime.h>

typedef __attribute__((ext_vector_type(16))) __bf16 bf16x16;
typedef __attribute__((ext_vector_type(8)))  __bf16 bf16x8;
typedef __attribute__((ext_vector_type(8)))  float  f32x8;
typedef __attribute__((ext_vector_type(4)))  float  f32x4;

__device__ __forceinline__ __bf16 f2bf(float x) { return (__bf16)x; }

__device__ __forceinline__ f32x8 zero8() {
  f32x8 z;
#pragma unroll
  for (int i = 0; i < 8; ++i) z[i] = 0.0f;
  return z;
}

// A fragment: 16x32 (MxK) bf16, row-major source with leading dim ld.
// lane: m = lane&15, h = lane>>4; VGPR0-3 hold K = h*8..h*8+7, VGPR4-7 hold +16.
__device__ __forceinline__ bf16x16 frag_a(const __bf16* base, int ld) {
  int lane = threadIdx.x & 31;
  const __bf16* r = base + (lane & 15) * ld + (lane >> 4) * 8;
  bf16x8 lo = *(const bf16x8*)(r);
  bf16x8 hi = *(const bf16x8*)(r + 16);
  bf16x16 a;
#pragma unroll
  for (int i = 0; i < 8; ++i) { a[i] = lo[i]; a[i + 8] = hi[i]; }
  return a;
}

// B fragment: 32x16 (KxN) bf16, source stored TRANSPOSED [n][k] with leading dim ld.
// lane: n = lane&15, h = lane>>4; VGPR0-3 hold K = h*8..h*8+7, VGPR4-7 hold +16.
__device__ __forceinline__ bf16x16 frag_b(const __bf16* baseT, int ld) {
  int lane = threadIdx.x & 31;
  const __bf16* r = baseT + (lane & 15) * ld + (lane >> 4) * 8;
  bf16x8 lo = *(const bf16x8*)(r);
  bf16x8 hi = *(const bf16x8*)(r + 16);
  bf16x16 b;
#pragma unroll
  for (int i = 0; i < 8; ++i) { b[i] = lo[i]; b[i + 8] = hi[i]; }
  return b;
}

__device__ __forceinline__ f32x8 wmma_bf16(bf16x16 a, bf16x16 b, f32x8 c) {
  return __builtin_amdgcn_wmma_f32_16x16x32_bf16(false, a, false, b, (short)0, c,
                                                 false, false);
}

// ---------------- Kernel 1: QKV GEMM (x[4096,1024] @ w[1024,3072] + b) -> bf16 qkv
// 64x64 block tile, K-step 64 (8 WMMAs/wave per barrier pair), prefetch next tile.
__global__ __launch_bounds__(128) void k_qkv_gemm(const float* __restrict__ x,
                                                  const float* __restrict__ w,
                                                  const float* __restrict__ bias,
                                                  __bf16* __restrict__ qkv) {
  __shared__ __align__(16) __bf16 sA[64 * 64];   // [m][k]
  __shared__ __align__(16) __bf16 sB[64 * 64];   // [n][k] transposed
  const int bn = blockIdx.x * 64;
  const int bm = blockIdx.y * 64;
  const int t = threadIdx.x;
  const int wid = t >> 5, lane = t & 31;
  const int wm = (wid >> 1) * 32, wn = (wid & 1) * 32;
  f32x8 c[2][2];
#pragma unroll
  for (int i = 0; i < 2; ++i)
#pragma unroll
    for (int j = 0; j < 2; ++j) c[i][j] = zero8();

  for (int k0 = 0; k0 < 1024; k0 += 64) {
    __syncthreads();
#pragma unroll
    for (int i = 0; i < 8; ++i) {               // A tile: 64x64 f32 -> bf16
      int lin = t + i * 128;                    // 1024 float4 chunks
      int m = lin >> 4, k4 = (lin & 15) * 4;
      f32x4 v = *(const f32x4*)(x + (size_t)(bm + m) * 1024 + k0 + k4);
#pragma unroll
      for (int j = 0; j < 4; ++j) sA[m * 64 + k4 + j] = f2bf(v[j]);
    }
#pragma unroll
    for (int i = 0; i < 8; ++i) {               // B tile: 64x64 f32, store [n][k]
      int lin = t + i * 128;
      int kk = lin >> 4, n4 = (lin & 15) * 4;
      f32x4 v = *(const f32x4*)(w + (size_t)(k0 + kk) * 3072 + bn + n4);
#pragma unroll
      for (int j = 0; j < 4; ++j) sB[(n4 + j) * 64 + kk] = f2bf(v[j]);
    }
    __syncthreads();
    if (k0 + 64 < 1024) {                       // prefetch next tiles into cache
      __builtin_prefetch(x + (size_t)(bm + (t >> 1)) * 1024 + k0 + 64 + (t & 1) * 32, 0, 1);
      __builtin_prefetch(w + (size_t)(k0 + 64 + (t >> 1)) * 3072 + bn + (t & 1) * 32, 0, 1);
    }
#pragma unroll
    for (int ks = 0; ks < 2; ++ks) {
      bf16x16 a0 = frag_a(sA + (wm + 0) * 64 + ks * 32, 64);
      bf16x16 a1 = frag_a(sA + (wm + 16) * 64 + ks * 32, 64);
      bf16x16 b0 = frag_b(sB + (wn + 0) * 64 + ks * 32, 64);
      bf16x16 b1 = frag_b(sB + (wn + 16) * 64 + ks * 32, 64);
      c[0][0] = wmma_bf16(a0, b0, c[0][0]);
      c[0][1] = wmma_bf16(a0, b1, c[0][1]);
      c[1][0] = wmma_bf16(a1, b0, c[1][0]);
      c[1][1] = wmma_bf16(a1, b1, c[1][1]);
    }
  }
  const int hl = lane >> 4, nn = lane & 15;
#pragma unroll
  for (int rs = 0; rs < 2; ++rs)
#pragma unroll
    for (int cs = 0; cs < 2; ++cs) {
      int col = bn + wn + cs * 16 + nn;
      float bvl = bias[col];
#pragma unroll
      for (int v = 0; v < 8; ++v) {
        int row = bm + wm + rs * 16 + hl * 8 + v;
        qkv[(size_t)row * 3072 + col] = f2bf(c[rs][cs][v] + bvl);
      }
    }
}

// ---------------- Kernel 2: RoPE(xpos) + layout: Q,K [B,H,N,64], V [B,H,64,N]
__global__ __launch_bounds__(256) void k_rope(const __bf16* __restrict__ qkv,
                                              __bf16* __restrict__ Q,
                                              __bf16* __restrict__ K,
                                              __bf16* __restrict__ V) {
  unsigned tid = blockIdx.x * 256u + threadIdx.x;   // 2^21 threads
  int p = tid & 31;
  int n = (tid >> 5) & 2047;
  int h = (tid >> 16) & 15;
  int b = tid >> 20;
  int gi = n >> 6, gj = n & 63;                     // spatial (32,64)
  int ap, tpos; float power;
  if (p < 16) { ap = p;      tpos = gi; power = (gi - 16) * (1.0f / 64.0f); }
  else        { ap = p - 16; tpos = gj; power = (gj - 32) * (1.0f / 64.0f); }
  float inv_freq = __powf(10000.0f, -(float)ap * (1.0f / 16.0f));
  float ang = (float)tpos * inv_freq;
  float cs = __cosf(ang), sn = __sinf(ang);
  float sbase = (2.0f * (float)ap + 12.8f) * (1.0f / 44.8f);
  float xs = __powf(sbase, power);
  int c0 = 2 * p;
  size_t base = ((size_t)(b * 2048 + n)) * 3072 + h * 64 + c0;
  float q0 = (float)qkv[base],        q1 = (float)qkv[base + 1];
  float k0 = (float)qkv[base + 1024], k1 = (float)qkv[base + 1025];
  float v0 = (float)qkv[base + 2048], v1 = (float)qkv[base + 2049];
  float qo0 = (q0 * cs - q1 * sn) * xs, qo1 = (q1 * cs + q0 * sn) * xs;
  float ixs = 1.0f / xs;
  float ko0 = (k0 * cs - k1 * sn) * ixs, ko1 = (k1 * cs + k0 * sn) * ixs;
  size_t bh = (size_t)(b * 16 + h);
  size_t qk = (bh * 2048 + n) * 64 + c0;
  Q[qk] = f2bf(qo0); Q[qk + 1] = f2bf(qo1);
  K[qk] = f2bf(ko0); K[qk + 1] = f2bf(ko1);
  size_t vb = (bh * 64 + c0) * 2048 + n;
  V[vb] = f2bf(v0); V[vb + 2048] = f2bf(v1);
}

// ---------------- Kernel 3: flash attention. block = (b,h, 64 q-rows), 4 waves.
__global__ __launch_bounds__(128) void k_attn(const __bf16* __restrict__ Q,
                                              const __bf16* __restrict__ K,
                                              const __bf16* __restrict__ V,
                                              __bf16* __restrict__ attn) {
  __shared__ __align__(16) __bf16 sK[32 * 64];      // [key][d]
  __shared__ __align__(16) __bf16 sV[64 * 32];      // [d][key] (transposed)
  __shared__ __align__(16) __bf16 sP[4][16 * 32];   // per-wave P scratch [row][key]
  const int bh = blockIdx.x >> 5;
  const int qb = blockIdx.x & 31;
  const int b = bh >> 4, h = bh & 15;
  const int t = threadIdx.x, wid = t >> 5, lane = t & 31;
  const int hl = lane >> 4, nn = lane & 15;
  const int qrow = qb * 64 + wid * 16;

  const __bf16* Qb = Q + ((size_t)bh * 2048 + qrow) * 64;
  bf16x16 aq0 = frag_a(Qb, 64);        // d 0..31
  bf16x16 aq1 = frag_a(Qb + 32, 64);   // d 32..63

  f32x8 o[4];
#pragma unroll
  for (int i = 0; i < 4; ++i) o[i] = zero8();
  float mi[8], li[8];
#pragma unroll
  for (int v = 0; v < 8; ++v) { mi[v] = -3.0e38f; li[v] = 0.0f; }

  for (int kt = 0; kt < 2048; kt += 32) {
    __syncthreads();
#pragma unroll
    for (int i = 0; i < 2; ++i) {                  // K tile [32][64]
      int ch = t + i * 128;                        // 256 chunks of 8 bf16
      int r = ch >> 3, c8 = (ch & 7) * 8;
      *(bf16x8*)(sK + r * 64 + c8) =
          *(const bf16x8*)(K + ((size_t)bh * 2048 + kt + r) * 64 + c8);
    }
#pragma unroll
    for (int i = 0; i < 2; ++i) {                  // V tile [64][32]
      int ch = t + i * 128;
      int d = ch >> 2, c8 = (ch & 3) * 8;
      *(bf16x8*)(sV + d * 32 + c8) =
          *(const bf16x8*)(V + ((size_t)bh * 64 + d) * 2048 + kt + c8);
    }
    __syncthreads();
    if (kt + 32 < 2048) {                          // prefetch next K/V tiles
      if (t < 32) __builtin_prefetch(K + ((size_t)bh * 2048 + kt + 32 + t) * 64, 0, 1);
      if (t < 64) __builtin_prefetch(V + ((size_t)bh * 64 + t) * 2048 + kt + 32, 0, 1);
    }

    f32x8 s[2];
#pragma unroll
    for (int kn = 0; kn < 2; ++kn) {               // S = Q @ K^T, 16 keys/subtile
      f32x8 z = zero8();
      bf16x16 bk0 = frag_b(sK + (kn * 16) * 64, 64);        // contraction d 0..31
      bf16x16 bk1 = frag_b(sK + (kn * 16) * 64 + 32, 64);   // contraction d 32..63
      z = wmma_bf16(aq0, bk0, z);
      z = wmma_bf16(aq1, bk1, z);
      s[kn] = z;
    }

#pragma unroll
    for (int v = 0; v < 8; ++v) {                  // online softmax per row
      float s0 = s[0][v] * 0.125f, s1 = s[1][v] * 0.125f;
      float mx = fmaxf(s0, s1);
#pragma unroll
      for (int msk = 1; msk < 16; msk <<= 1) mx = fmaxf(mx, __shfl_xor(mx, msk, 32));
      float mnew = fmaxf(mi[v], mx);
      float p0 = __expf(s0 - mnew), p1 = __expf(s1 - mnew);
      s[0][v] = p0; s[1][v] = p1;
      float rsum = p0 + p1;
#pragma unroll
      for (int msk = 1; msk < 16; msk <<= 1) rsum += __shfl_xor(rsum, msk, 32);
      float corr = __expf(mi[v] - mnew);
      li[v] = li[v] * corr + rsum;
      mi[v] = mnew;
#pragma unroll
      for (int ss = 0; ss < 4; ++ss) o[ss][v] *= corr;
    }

    // C-layout -> A-layout transpose of P through per-wave LDS scratch
    __bf16* pw = &sP[wid][0];
#pragma unroll
    for (int kn = 0; kn < 2; ++kn)
#pragma unroll
      for (int v = 0; v < 8; ++v)
        pw[(hl * 8 + v) * 32 + kn * 16 + nn] = f2bf(s[kn][v]);
    __builtin_amdgcn_wave_barrier();
    asm volatile("s_wait_dscnt 0" ::: "memory");
    bf16x16 pa = frag_a(pw, 32);

#pragma unroll
    for (int ss = 0; ss < 4; ++ss) {               // O += P @ V
      bf16x16 bvv = frag_b(sV + (ss * 16) * 32, 32);
      o[ss] = wmma_bf16(pa, bvv, o[ss]);
    }
  }

#pragma unroll
  for (int ss = 0; ss < 4; ++ss)
#pragma unroll
    for (int v = 0; v < 8; ++v) {
      int row = qrow + hl * 8 + v;
      int col = h * 64 + ss * 16 + nn;
      attn[((size_t)b * 2048 + row) * 1024 + col] = f2bf(o[ss][v] / li[v]);
    }
}

// ---------------- Kernel 4: out = attn[4096,1024](bf16) @ w_proj[1024,1024] + b
__global__ __launch_bounds__(128) void k_proj_gemm(const __bf16* __restrict__ A,
                                                   const float* __restrict__ w,
                                                   const float* __restrict__ bias,
                                                   float* __restrict__ out) {
  __shared__ __align__(16) __bf16 sA[64 * 64];
  __shared__ __align__(16) __bf16 sB[64 * 64];
  const int bn = blockIdx.x * 64;
  const int bm = blockIdx.y * 64;
  const int t = threadIdx.x;
  const int wid = t >> 5, lane = t & 31;
  const int wm = (wid >> 1) * 32, wn = (wid & 1) * 32;
  f32x8 c[2][2];
#pragma unroll
  for (int i = 0; i < 2; ++i)
#pragma unroll
    for (int j = 0; j < 2; ++j) c[i][j] = zero8();

  for (int k0 = 0; k0 < 1024; k0 += 64) {
    __syncthreads();
#pragma unroll
    for (int i = 0; i < 4; ++i) {                  // A tile bf16 64x64
      int ch = t + i * 128;                        // 512 chunks of 8 bf16
      int m = ch >> 3, c8 = (ch & 7) * 8;
      *(bf16x8*)(sA + m * 64 + c8) =
          *(const bf16x8*)(A + (size_t)(bm + m) * 1024 + k0 + c8);
    }
#pragma unroll
    for (int i = 0; i < 8; ++i) {                  // B tile f32 64x64 -> [n][k]
      int lin = t + i * 128;
      int kk = lin >> 4, n4 = (lin & 15) * 4;
      f32x4 v = *(const f32x4*)(w + (size_t)(k0 + kk) * 1024 + bn + n4);
#pragma unroll
      for (int j = 0; j < 4; ++j) sB[(n4 + j) * 64 + kk] = f2bf(v[j]);
    }
    __syncthreads();
    if (k0 + 64 < 1024) {
      if (t < 64) __builtin_prefetch(A + (size_t)(bm + t) * 1024 + k0 + 64, 0, 1);
      __builtin_prefetch(w + (size_t)(k0 + 64 + (t >> 1)) * 1024 + bn + (t & 1) * 32, 0, 1);
    }
#pragma unroll
    for (int ks = 0; ks < 2; ++ks) {
      bf16x16 a0 = frag_a(sA + (wm + 0) * 64 + ks * 32, 64);
      bf16x16 a1 = frag_a(sA + (wm + 16) * 64 + ks * 32, 64);
      bf16x16 b0 = frag_b(sB + (wn + 0) * 64 + ks * 32, 64);
      bf16x16 b1 = frag_b(sB + (wn + 16) * 64 + ks * 32, 64);
      c[0][0] = wmma_bf16(a0, b0, c[0][0]);
      c[0][1] = wmma_bf16(a0, b1, c[0][1]);
      c[1][0] = wmma_bf16(a1, b0, c[1][0]);
      c[1][1] = wmma_bf16(a1, b1, c[1][1]);
    }
  }
  const int hl = lane >> 4, nn = lane & 15;
#pragma unroll
  for (int rs = 0; rs < 2; ++rs)
#pragma unroll
    for (int cs = 0; cs < 2; ++cs) {
      int col = bn + wn + cs * 16 + nn;
      float bvl = bias[col];
#pragma unroll
      for (int v = 0; v < 8; ++v) {
        int row = bm + wm + rs * 16 + hl * 8 + v;
        out[(size_t)row * 1024 + col] = c[rs][cs][v] + bvl;
      }
    }
}

extern "C" void kernel_launch(void* const* d_in, const int* in_sizes, int n_in,
                              void* d_out, int out_size, void* d_ws, size_t ws_size,
                              hipStream_t stream) {
  const float* x      = (const float*)d_in[0];
  const float* w_qkv  = (const float*)d_in[1];
  const float* b_qkv  = (const float*)d_in[2];
  const float* w_proj = (const float*)d_in[3];
  const float* b_proj = (const float*)d_in[4];
  float* out = (float*)d_out;

  char* ws = (char*)d_ws;
  __bf16* qkv  = (__bf16*)ws;                       // 4096*3072 bf16 = 24 MB
  __bf16* Q    = (__bf16*)(ws + (size_t)25165824);  // 2*16*2048*64
  __bf16* Kp   = Q + (size_t)4194304;
  __bf16* Vp   = Kp + (size_t)4194304;
  __bf16* attn = Vp + (size_t)4194304;              // 4096*1024

  k_qkv_gemm<<<dim3(48, 64), 128, 0, stream>>>(x, w_qkv, b_qkv, qkv);
  k_rope<<<8192, 256, 0, stream>>>(qkv, Q, Kp, Vp);
  k_attn<<<1024, 128, 0, stream>>>(Q, Kp, Vp, attn);
  k_proj_gemm<<<dim3(16, 64), 128, 0, stream>>>(attn, w_proj, b_proj, out);
}
